// PointTransformer_78426102825596
// MI455X (gfx1250) — matrix operand order repaired and checked by reading, not compiled, not run on hardware
//
#include <hip/hip_runtime.h>

// ---------------------------------------------------------------------------
// Point Transformer block for MI455X (gfx1250), wave32 + bf16 WMMA.
// Shapes: B=2, CIN=64, C=128, N=8192, K=16, GROUPS=8.
// ---------------------------------------------------------------------------

#define BB   2
#define NNP  8192
#define KKN  16
#define CC   128
#define CIN_ 64
#define GG   8
#define CG   16
#define SSP  (NNP * KKN)   // 131072 columns per batch

typedef unsigned int uint;
typedef unsigned short ushort;
typedef __attribute__((ext_vector_type(16))) __bf16 v16bf;
typedef __attribute__((ext_vector_type(8)))  float  v8f;

union BF16Frag { v16bf v; uint4 q[2]; };

__device__ __forceinline__ ushort f2bf(float f) {
    union { float f; uint u; } x; x.f = f;
    uint u = x.u;
    u += 0x7FFFu + ((u >> 16) & 1u);          // round-to-nearest-even
    return (ushort)(u >> 16);
}
__device__ __forceinline__ uint pack2(float a, float b) {
    return (uint)f2bf(a) | ((uint)f2bf(b) << 16);
}
__device__ __forceinline__ float lrelu(float x) { return x >= 0.0f ? x : 0.1f * x; }
__device__ __forceinline__ v8f zero8() {
    v8f z = {0.f, 0.f, 0.f, 0.f, 0.f, 0.f, 0.f, 0.f};
    return z;
}
__device__ __forceinline__ v8f wmma_bf16(const BF16Frag& a, const BF16Frag& b, v8f c) {
    return __builtin_amdgcn_wmma_f32_16x16x32_bf16(false, a.v, false, b.v,
                                                   (short)0, c, false, false);
}

// A fragment: 16x32 bf16. lanes 0-15: m=lane, K halves {k0..k0+7, k0+16..k0+23};
// lanes 16-31: m=lane-16, K halves {k0+8..k0+15, k0+24..k0+31}.
__device__ __forceinline__ void load_afrag(const ushort* W, int lane, int k0, int ld,
                                           BF16Frag& f) {
    int m  = lane & 15;
    int ho = (lane >> 4) << 3;                 // 0 or 8
    const ushort* base = W + m * ld + k0 + ho;
    f.q[0] = *(const uint4*)(base);
    f.q[1] = *(const uint4*)(base + 16);
}
// B fragment: 32x16 bf16, X stored [col][channel] contiguous.
// lanes 0-15: n=lane, K {k0..k0+7, k0+16..k0+23}; lanes 16-31: n=lane-16, K {k0+8..k0+15, k0+24..k0+31}.
__device__ __forceinline__ void load_bfrag(const ushort* X, int lane, int colbase,
                                           int k0, int ld, BF16Frag& f) {
    int n  = lane & 15;
    int kh = (lane >> 4) << 3;                 // 0 or 8
    const ushort* base = X + (colbase + n) * ld + k0 + kh;
    f.q[0] = *(const uint4*)(base);
    f.q[1] = *(const uint4*)(base + 16);
}

// ---------------------------------------------------------------------------
// Kernel 1: brute force KNN (top-16 smallest d2 = sq[n]+sq[m]-2*dot)
// grid = B * N/128, block = 128
// ---------------------------------------------------------------------------
__global__ __launch_bounds__(128) void k_knn(const float* __restrict__ xyz,
                                             int* __restrict__ idxout) {
    int b    = blockIdx.x >> 6;
    int tile = blockIdx.x & 63;
    int pq   = tile * 128 + threadIdx.x;
    const float* xb = xyz + (size_t)b * 3 * NNP;
    float qx = xb[pq], qy = xb[NNP + pq], qz = xb[2 * NNP + pq];
    float sqq = qx * qx + qy * qy + qz * qz;

    float nd[KKN]; int ni[KKN];
#pragma unroll
    for (int i = 0; i < KKN; i++) { nd[i] = 3.4e38f; ni[i] = 0; }
    float wd = 3.4e38f; int wsl = 0;

    __shared__ float cx[256], cy[256], cz[256], cs[256];
    for (int t0 = 0; t0 < NNP; t0 += 256) {
        __syncthreads();
        for (int j = threadIdx.x; j < 256; j += 128) {
            float x = xb[t0 + j], y = xb[NNP + t0 + j], z = xb[2 * NNP + t0 + j];
            cx[j] = x; cy[j] = y; cz[j] = z; cs[j] = x * x + y * y + z * z;
        }
        __syncthreads();
        for (int j = 0; j < 256; j++) {
            float d = sqq + cs[j] - 2.0f * (qx * cx[j] + qy * cy[j] + qz * cz[j]);
            if (d < wd) {
#pragma unroll
                for (int i = 0; i < KKN; i++)
                    if (i == wsl) { nd[i] = d; ni[i] = t0 + j; }
                wd = nd[0]; wsl = 0;
#pragma unroll
                for (int i = 1; i < KKN; i++)
                    if (nd[i] > wd) { wd = nd[i]; wsl = i; }
            }
        }
    }
    int* op = idxout + ((size_t)b * NNP + pq) * KKN;
#pragma unroll
    for (int i = 0; i < KKN; i++) op[i] = ni[i];
}

// ---------------------------------------------------------------------------
// Shared helper: 128x128 GEMM strip (per wave: 16 out channels x 64 cols),
// bias add, store f32 column-major [n][c].
// ---------------------------------------------------------------------------
__device__ __forceinline__ void gemm128_store(const ushort* Wl, const ushort* Xl,
                                              int c0, int lane,
                                              const float* __restrict__ bias,
                                              float* __restrict__ out0) {
    BF16Frag a[4];
#pragma unroll
    for (int ks = 0; ks < 4; ks++) load_afrag(Wl, lane, ks * 32, 136, a[ks]);
    int hi = lane >> 4, ncol = lane & 15;
    float4 b0 = *(const float4*)&bias[c0 + 8 * hi];
    float4 b1 = *(const float4*)&bias[c0 + 8 * hi + 4];
#pragma unroll
    for (int st = 0; st < 4; st++) {
        v8f acc = zero8();
#pragma unroll
        for (int ks = 0; ks < 4; ks++) {
            BF16Frag bf; load_bfrag(Xl, lane, st * 16, ks * 32, 136, bf);
            acc = wmma_bf16(a[ks], bf, acc);
        }
        float* dst = out0 + (size_t)(st * 16 + ncol) * CC + c0 + 8 * hi;
        *(float4*)dst       = make_float4(acc[0] + b0.x, acc[1] + b0.y, acc[2] + b0.z, acc[3] + b0.w);
        *(float4*)(dst + 4) = make_float4(acc[4] + b1.x, acc[5] + b1.y, acc[6] + b1.z, acc[7] + b1.w);
    }
}

// ---------------------------------------------------------------------------
// Kernel 2: new_feat = pre_w@feat+b, then q/k/v GEMMs from LDS-resident new_feat.
// grid = B * N/64, block = 256 (8 waves, one 16-channel strip each)
// ---------------------------------------------------------------------------
__global__ __launch_bounds__(256) void k_pre_qkv(
    const float* __restrict__ feat,
    const float* __restrict__ pre_w, const float* __restrict__ pre_b,
    const float* __restrict__ wq_w,  const float* __restrict__ wq_b,
    const float* __restrict__ wk_w,  const float* __restrict__ wk_b,
    const float* __restrict__ wv_w,  const float* __restrict__ wv_b,
    float* __restrict__ nf, float* __restrict__ qb,
    float* __restrict__ kb, float* __restrict__ vb) {
    const int LDW = 136, LDX = 72;
    __shared__ __align__(16) ushort Wl[CC * 136];     // 34.0 KiB
    __shared__ __align__(16) ushort Xf[64 * 72];      //  9.0 KiB
    __shared__ __align__(16) ushort NFl[64 * 136];    // 17.0 KiB
    int b  = blockIdx.x >> 7;
    int p0 = (blockIdx.x & 127) * 64;
    int tid = threadIdx.x, lane = tid & 31, wid = tid >> 5;

    for (int i = tid; i < CIN_ * 64; i += 256) {
        int c = i >> 6, col = i & 63;
        Xf[col * LDX + c] = f2bf(feat[((size_t)b * CIN_ + c) * NNP + p0 + col]);
    }
    for (int i = tid; i < CC * CIN_; i += 256)
        Wl[(i >> 6) * LDW + (i & 63)] = f2bf(pre_w[i]);
    __syncthreads();

    int c0 = wid * 16, hi = lane >> 4, ncol = lane & 15;
    {   // pre GEMM: K = 64 (2 k-steps)
        BF16Frag a0, a1;
        load_afrag(Wl, lane, 0, LDW, a0);
        load_afrag(Wl, lane, 32, LDW, a1);
        float4 pb0 = *(const float4*)&pre_b[c0 + 8 * hi];
        float4 pb1 = *(const float4*)&pre_b[c0 + 8 * hi + 4];
        float* nf0 = nf + ((size_t)b * NNP + p0) * CC;
#pragma unroll
        for (int st = 0; st < 4; st++) {
            v8f acc = zero8();
            BF16Frag bf;
            load_bfrag(Xf, lane, st * 16, 0, LDX, bf);  acc = wmma_bf16(a0, bf, acc);
            load_bfrag(Xf, lane, st * 16, 32, LDX, bf); acc = wmma_bf16(a1, bf, acc);
            float v0 = acc[0] + pb0.x, v1 = acc[1] + pb0.y, v2 = acc[2] + pb0.z, v3 = acc[3] + pb0.w;
            float v4 = acc[4] + pb1.x, v5 = acc[5] + pb1.y, v6 = acc[6] + pb1.z, v7 = acc[7] + pb1.w;
            int col = st * 16 + ncol;
            float* dst = nf0 + (size_t)col * CC + c0 + 8 * hi;
            *(float4*)dst       = make_float4(v0, v1, v2, v3);
            *(float4*)(dst + 4) = make_float4(v4, v5, v6, v7);
            uint4 pk;
            pk.x = pack2(v0, v1); pk.y = pack2(v2, v3);
            pk.z = pack2(v4, v5); pk.w = pack2(v6, v7);
            *(uint4*)&NFl[col * LDW + c0 + 8 * hi] = pk;
        }
    }
    __syncthreads();
    for (int i = tid; i < CC * CC; i += 256) Wl[(i >> 7) * LDW + (i & 127)] = f2bf(wq_w[i]);
    __syncthreads();
    gemm128_store(Wl, NFl, c0, lane, wq_b, qb + ((size_t)b * NNP + p0) * CC);
    __syncthreads();
    for (int i = tid; i < CC * CC; i += 256) Wl[(i >> 7) * LDW + (i & 127)] = f2bf(wk_w[i]);
    __syncthreads();
    gemm128_store(Wl, NFl, c0, lane, wk_b, kb + ((size_t)b * NNP + p0) * CC);
    __syncthreads();
    for (int i = tid; i < CC * CC; i += 256) Wl[(i >> 7) * LDW + (i & 127)] = f2bf(wv_w[i]);
    __syncthreads();
    gemm128_store(Wl, NFl, c0, lane, wv_b, vb + ((size_t)b * NNP + p0) * CC);
}

// ---------------------------------------------------------------------------
// Kernel 3: GroupNorm statistics of pos1 output (recomputed, never stored).
// grid = B * N/128, block = 256
// ---------------------------------------------------------------------------
__global__ __launch_bounds__(256) void k_pos1_stats(
    const float* __restrict__ xyz, const int* __restrict__ idx,
    const float* __restrict__ pos1_w, const float* __restrict__ pos1_b,
    float* __restrict__ part) {
    __shared__ float w0[CC], w1[CC], w2[CC], bbv[CC];
    __shared__ float wred[8];
    int b  = blockIdx.x >> 6;
    int p0 = (blockIdx.x & 63) * 128;
    int tid = threadIdx.x;
    for (int i = tid; i < CC; i += 256) {
        w0[i] = pos1_w[i * 3]; w1[i] = pos1_w[i * 3 + 1];
        w2[i] = pos1_w[i * 3 + 2]; bbv[i] = pos1_b[i];
    }
    __syncthreads();
    const float* xb = xyz + (size_t)b * 3 * NNP;
    float gs[GG], gss[GG];
#pragma unroll
    for (int g = 0; g < GG; g++) { gs[g] = 0.f; gss[g] = 0.f; }
    for (int pi = tid; pi < 128 * KKN; pi += 256) {
        int p = p0 + (pi >> 4);
        int j = idx[((size_t)b * NNP + p) * KKN + (pi & 15)];
        float rx = xb[j] - xb[p];
        float ry = xb[NNP + j] - xb[NNP + p];
        float rz = xb[2 * NNP + j] - xb[2 * NNP + p];
#pragma unroll
        for (int g = 0; g < GG; g++) {
            float s = 0.f, ss = 0.f;
#pragma unroll
            for (int cc = 0; cc < CG; cc++) {
                int c = g * CG + cc;
                float v = fmaf(w0[c], rx, fmaf(w1[c], ry, fmaf(w2[c], rz, bbv[c])));
                s += v; ss += v * v;
            }
            gs[g] += s; gss[g] += ss;
        }
    }
    int lane = tid & 31, wid = tid >> 5;
    for (int g = 0; g < GG; g++) {
        float s = gs[g], ss = gss[g];
        for (int o = 16; o; o >>= 1) { s += __shfl_xor(s, o, 32); ss += __shfl_xor(ss, o, 32); }
        if (lane == 0) wred[wid] = s;
        __syncthreads();
        if (tid == 0) { float t = 0; for (int w = 0; w < 8; w++) t += wred[w];
                        part[((size_t)blockIdx.x * GG + g) * 2] = t; }
        __syncthreads();
        if (lane == 0) wred[wid] = ss;
        __syncthreads();
        if (tid == 0) { float t = 0; for (int w = 0; w < 8; w++) t += wred[w];
                        part[((size_t)blockIdx.x * GG + g) * 2 + 1] = t; }
        __syncthreads();
    }
}

// ---------------------------------------------------------------------------
// GroupNorm partial reduction -> per (b,c) scale/shift.  grid = B*G, block=256
// ---------------------------------------------------------------------------
__global__ __launch_bounds__(256) void k_gn_reduce(
    const float* __restrict__ part, int nblk, float cnt,
    const float* __restrict__ gamma, const float* __restrict__ beta,
    float* __restrict__ gnp) {
    __shared__ float wr1[8], wr2[8], mv[2];
    int b = blockIdx.x >> 3, g = blockIdx.x & 7;
    int tid = threadIdx.x;
    float s1 = 0.f, s2 = 0.f;
    for (int i = tid; i < nblk; i += 256) {
        size_t o = (((size_t)b * nblk + i) * GG + g) * 2;
        s1 += part[o]; s2 += part[o + 1];
    }
    for (int o = 16; o; o >>= 1) { s1 += __shfl_xor(s1, o, 32); s2 += __shfl_xor(s2, o, 32); }
    if ((tid & 31) == 0) { wr1[tid >> 5] = s1; wr2[tid >> 5] = s2; }
    __syncthreads();
    if (tid == 0) {
        float a = 0, c = 0;
        for (int w = 0; w < 8; w++) { a += wr1[w]; c += wr2[w]; }
        float mean = a / cnt;
        float var = c / cnt - mean * mean;
        mv[0] = mean; mv[1] = rsqrtf(var + 1e-5f);
    }
    __syncthreads();
    if (tid < CG) {
        int c = g * CG + tid;
        float sc = gamma[c] * mv[1];
        float sh = beta[c] - mv[0] * sc;
        gnp[((size_t)b * CC + c) * 2]     = sc;
        gnp[((size_t)b * CC + c) * 2 + 1] = sh;
    }
}

// ---------------------------------------------------------------------------
// Kernel 4 (heavy): pos1->GN->lrelu->[WMMA pos2]->(q - k_gather + pos)->
// [WMMA att1] -> t (pre-GN) + group stats.   grid = B*N/4, block = 256
// ---------------------------------------------------------------------------
__global__ __launch_bounds__(256) void k_mid(
    const float* __restrict__ xyz, const int* __restrict__ idx,
    const float* __restrict__ pos1_w, const float* __restrict__ pos1_b,
    const float* __restrict__ gnp1,
    const float* __restrict__ pos2_w, const float* __restrict__ pos2_b,
    const float* __restrict__ att1_w, const float* __restrict__ att1_b,
    const float* __restrict__ qb, const float* __restrict__ kb,
    float* __restrict__ tbuf, float* __restrict__ part) {
    const int LDW = 136, LDX = 136;
    __shared__ __align__(16) ushort Wl[CC * 136];
    __shared__ __align__(16) ushort Xl[64 * 136];
    __shared__ int   idl[64];
    __shared__ float w0[CC], w1[CC], w2[CC], bbv[CC], sc1[CC], sh1[CC];
    int b  = blockIdx.x >> 11;
    int p0 = (blockIdx.x & 2047) * 4;
    int tid = threadIdx.x, lane = tid & 31, wid = tid >> 5;

    for (int i = tid; i < CC; i += 256) {
        w0[i] = pos1_w[i * 3]; w1[i] = pos1_w[i * 3 + 1];
        w2[i] = pos1_w[i * 3 + 2]; bbv[i] = pos1_b[i];
        sc1[i] = gnp1[((size_t)b * CC + i) * 2];
        sh1[i] = gnp1[((size_t)b * CC + i) * 2 + 1];
    }
    for (int i = tid; i < 64; i += 256)
        idl[i] = idx[((size_t)b * NNP + p0 + (i >> 4)) * KKN + (i & 15)];
    for (int i = tid; i < CC * CC; i += 256)
        Wl[(i >> 7) * LDW + (i & 127)] = f2bf(pos2_w[i]);
    __syncthreads();

    // Build X1 = bf16(lrelu(gn1(pos1(rel)))) : 4 threads/col x 32 ch
    {
        const float* xb = xyz + (size_t)b * 3 * NNP;
        int col = tid >> 2, cb = (tid & 3) * 32;
        int p = p0 + (col >> 4);
        int j = idl[col];
        float rx = xb[j] - xb[p], ry = xb[NNP + j] - xb[NNP + p], rz = xb[2 * NNP + j] - xb[2 * NNP + p];
#pragma unroll
        for (int cc = 0; cc < 32; cc += 2) {
            int c = cb + cc;
            float v0 = fmaf(w0[c], rx, fmaf(w1[c], ry, fmaf(w2[c], rz, bbv[c])));
            float v1 = fmaf(w0[c + 1], rx, fmaf(w1[c + 1], ry, fmaf(w2[c + 1], rz, bbv[c + 1])));
            v0 = lrelu(fmaf(v0, sc1[c], sh1[c]));
            v1 = lrelu(fmaf(v1, sc1[c + 1], sh1[c + 1]));
            *(uint*)&Xl[col * LDX + c] = pack2(v0, v1);
        }
    }
    __syncthreads();

    int c0 = wid * 16, hi = lane >> 4, ncol = lane & 15;
    BF16Frag a[4];
#pragma unroll
    for (int ks = 0; ks < 4; ks++) load_afrag(Wl, lane, ks * 32, LDW, a[ks]);
    v8f P[4];
#pragma unroll
    for (int st = 0; st < 4; st++) {
        v8f acc = zero8();
#pragma unroll
        for (int ks = 0; ks < 4; ks++) {
            BF16Frag bf; load_bfrag(Xl, lane, st * 16, ks * 32, LDX, bf);
            acc = wmma_bf16(a[ks], bf, acc);
        }
        P[st] = acc;
    }
    __syncthreads();   // all pos2 GEMMs done -> X may be overwritten

    {   // att_in = q - k_gather + pos2 + pos2_b  -> bf16 -> X
        float4 pb0 = *(const float4*)&pos2_b[c0 + 8 * hi];
        float4 pb1 = *(const float4*)&pos2_b[c0 + 8 * hi + 4];
#pragma unroll
        for (int st = 0; st < 4; st++) {
            int p = p0 + st;
            int j = idl[st * 16 + ncol];
            const float* qp = qb + ((size_t)b * NNP + p) * CC + c0 + 8 * hi;
            const float* kp = kb + ((size_t)b * NNP + j) * CC + c0 + 8 * hi;
            float4 q0 = *(const float4*)qp, q1 = *(const float4*)(qp + 4);
            float4 k0 = *(const float4*)kp, k1 = *(const float4*)(kp + 4);
            v8f acc = P[st];
            float v0 = q0.x - k0.x + acc[0] + pb0.x;
            float v1 = q0.y - k0.y + acc[1] + pb0.y;
            float v2 = q0.z - k0.z + acc[2] + pb0.z;
            float v3 = q0.w - k0.w + acc[3] + pb0.w;
            float v4 = q1.x - k1.x + acc[4] + pb1.x;
            float v5 = q1.y - k1.y + acc[5] + pb1.y;
            float v6 = q1.z - k1.z + acc[6] + pb1.z;
            float v7 = q1.w - k1.w + acc[7] + pb1.w;
            uint4 pk;
            pk.x = pack2(v0, v1); pk.y = pack2(v2, v3);
            pk.z = pack2(v4, v5); pk.w = pack2(v6, v7);
            *(uint4*)&Xl[(size_t)(st * 16 + ncol) * LDX + c0 + 8 * hi] = pk;
        }
    }
    __syncthreads();
    for (int i = tid; i < CC * CC; i += 256)
        Wl[(i >> 7) * LDW + (i & 127)] = f2bf(att1_w[i]);
    __syncthreads();

#pragma unroll
    for (int ks = 0; ks < 4; ks++) load_afrag(Wl, lane, ks * 32, LDW, a[ks]);
    float s1 = 0.f, s2 = 0.f;
    float4 ab0 = *(const float4*)&att1_b[c0 + 8 * hi];
    float4 ab1 = *(const float4*)&att1_b[c0 + 8 * hi + 4];
    float* t0 = tbuf + ((size_t)b * SSP + (size_t)p0 * KKN) * CC;
#pragma unroll
    for (int st = 0; st < 4; st++) {
        v8f acc = zero8();
#pragma unroll
        for (int ks = 0; ks < 4; ks++) {
            BF16Frag bf; load_bfrag(Xl, lane, st * 16, ks * 32, LDX, bf);
            acc = wmma_bf16(a[ks], bf, acc);
        }
        float v[8];
        v[0] = acc[0] + ab0.x; v[1] = acc[1] + ab0.y; v[2] = acc[2] + ab0.z; v[3] = acc[3] + ab0.w;
        v[4] = acc[4] + ab1.x; v[5] = acc[5] + ab1.y; v[6] = acc[6] + ab1.z; v[7] = acc[7] + ab1.w;
        float* dst = t0 + (size_t)(st * 16 + ncol) * CC + c0 + 8 * hi;
        *(float4*)dst       = make_float4(v[0], v[1], v[2], v[3]);
        *(float4*)(dst + 4) = make_float4(v[4], v[5], v[6], v[7]);
#pragma unroll
        for (int r = 0; r < 8; r++) { s1 += v[r]; s2 += v[r] * v[r]; }
    }
    // wave == one GN group (channels [16*wid, 16*wid+16))
    for (int o = 16; o; o >>= 1) { s1 += __shfl_xor(s1, o, 32); s2 += __shfl_xor(s2, o, 32); }
    if (lane == 0) {
        size_t o = ((size_t)blockIdx.x * GG + wid) * 2;
        part[o] = s1; part[o + 1] = s2;
    }
}

// ---------------------------------------------------------------------------
// Kernel 5: GN->lrelu->[WMMA att2]->softmax(K)->sum(att*v_gather)+new_feat
// -> scalar post conv -> z (pre-GN) + stats.   grid = B*N/4, block = 256
// ---------------------------------------------------------------------------
__global__ __launch_bounds__(256) void k_final(
    const int* __restrict__ idx,
    const float* __restrict__ tbuf, const float* __restrict__ gnp2,
    const float* __restrict__ att2_w, const float* __restrict__ att2_b,
    const float* __restrict__ vb, const float* __restrict__ nf,
    const float* __restrict__ post_w, const float* __restrict__ post_b,
    float* __restrict__ zbuf, float* __restrict__ part) {
    const int LDW = 136, LDX = 136;
    __shared__ __align__(16) ushort Wl[CC * 136];
    __shared__ __align__(16) ushort Xl[64 * 136];
    __shared__ __align__(16) float Y[4][CC];
    __shared__ int idl[64];
    __shared__ float sc2[CC], sh2[CC];
    int b  = blockIdx.x >> 11;
    int p0 = (blockIdx.x & 2047) * 4;
    int tid = threadIdx.x, lane = tid & 31, wid = tid >> 5;

    for (int i = tid; i < CC; i += 256) {
        sc2[i] = gnp2[((size_t)b * CC + i) * 2];
        sh2[i] = gnp2[((size_t)b * CC + i) * 2 + 1];
    }
    for (int i = tid; i < 64; i += 256)
        idl[i] = idx[((size_t)b * NNP + p0 + (i >> 4)) * KKN + (i & 15)];
    for (int i = tid; i < CC * CC; i += 256)
        Wl[(i >> 7) * LDW + (i & 127)] = f2bf(att2_w[i]);
    __syncthreads();   // sc2 visible for stage below

    {   // X = bf16(lrelu(gn2(t))) : 4 threads/col x 32 ch
        int col = tid >> 2, cb = (tid & 3) * 32;
        const float* src = tbuf + ((size_t)b * SSP + (size_t)p0 * KKN + col) * CC + cb;
#pragma unroll
        for (int cc = 0; cc < 32; cc += 2) {
            int c = cb + cc;
            float v0 = lrelu(fmaf(src[cc], sc2[c], sh2[c]));
            float v1 = lrelu(fmaf(src[cc + 1], sc2[c + 1], sh2[c + 1]));
            *(uint*)&Xl[col * LDX + c] = pack2(v0, v1);
        }
    }
    __syncthreads();

    int c0 = wid * 16, hi = lane >> 4, ncol = lane & 15;
    BF16Frag a[4];
#pragma unroll
    for (int ks = 0; ks < 4; ks++) load_afrag(Wl, lane, ks * 32, LDW, a[ks]);
    float4 ab0 = *(const float4*)&att2_b[c0 + 8 * hi];
    float4 ab1 = *(const float4*)&att2_b[c0 + 8 * hi + 4];
#pragma unroll
    for (int st = 0; st < 4; st++) {
        v8f acc = zero8();
#pragma unroll
        for (int ks = 0; ks < 4; ks++) {
            BF16Frag bf; load_bfrag(Xl, lane, st * 16, ks * 32, LDX, bf);
            acc = wmma_bf16(a[ks], bf, acc);
        }
        float e[8];
        e[0] = acc[0] + ab0.x; e[1] = acc[1] + ab0.y; e[2] = acc[2] + ab0.z; e[3] = acc[3] + ab0.w;
        e[4] = acc[4] + ab1.x; e[5] = acc[5] + ab1.y; e[6] = acc[6] + ab1.z; e[7] = acc[7] + ab1.w;
        // softmax over K: 16-lane butterflies within each wave half
#pragma unroll
        for (int r = 0; r < 8; r++) {
            float m = e[r];
            for (int o = 8; o; o >>= 1) m = fmaxf(m, __shfl_xor(m, o, 16));
            float ex = expf(e[r] - m);
            float s = ex;
            for (int o = 8; o; o >>= 1) s += __shfl_xor(s, o, 16);
            e[r] = ex / s;
        }
        // weighted v gather + K-reduction
        int j = idl[st * 16 + ncol];
        const float* vp = vb + ((size_t)b * NNP + j) * CC + c0 + 8 * hi;
        float4 v0 = *(const float4*)vp, v1 = *(const float4*)(vp + 4);
        float ov[8];
        ov[0] = e[0] * v0.x; ov[1] = e[1] * v0.y; ov[2] = e[2] * v0.z; ov[3] = e[3] * v0.w;
        ov[4] = e[4] * v1.x; ov[5] = e[5] * v1.y; ov[6] = e[6] * v1.z; ov[7] = e[7] * v1.w;
#pragma unroll
        for (int r = 0; r < 8; r++) {
            float s = ov[r];
            for (int o = 8; o; o >>= 1) s += __shfl_xor(s, o, 16);
            ov[r] = s;
        }
        if (ncol == 0) {
            *(float4*)&Y[st][c0 + 8 * hi]     = make_float4(ov[0], ov[1], ov[2], ov[3]);
            *(float4*)&Y[st][c0 + 8 * hi + 4] = make_float4(ov[4], ov[5], ov[6], ov[7]);
        }
    }
    __syncthreads();
    // y = out + new_feat
    for (int i = tid; i < 4 * CC; i += 256) {
        int p = i >> 7, c = i & 127;
        Y[p][c] += nf[((size_t)b * NNP + p0 + p) * CC + c];
    }
    __syncthreads();
    // post conv (scalar, tiny) + stats; threads 32w..32w+31 all have c in group w
    {
        int c = tid >> 1;
        int g = c >> 4;
        const float* wr = post_w + (size_t)c * CC;
        float pb = post_b[c];
        float s1 = 0.f, s2 = 0.f;
#pragma unroll
        for (int pp = 0; pp < 2; pp++) {
            int p = (tid & 1) * 2 + pp;
            float acc = pb;
            for (int cc = 0; cc < CC; cc++) acc = fmaf(wr[cc], Y[p][cc], acc);
            zbuf[((size_t)b * CC + c) * NNP + p0 + p] = acc;
            s1 += acc; s2 += acc * acc;
        }
        for (int o = 16; o; o >>= 1) { s1 += __shfl_xor(s1, o, 32); s2 += __shfl_xor(s2, o, 32); }
        if ((tid & 31) == 0) {
            size_t o = ((size_t)blockIdx.x * GG + g) * 2;
            part[o] = s1; part[o + 1] = s2;
        }
    }
}

// ---------------------------------------------------------------------------
// Kernel 6: final GN + lrelu.   grid = B*C*N/256
// ---------------------------------------------------------------------------
__global__ __launch_bounds__(256) void k_finalize(const float* __restrict__ zbuf,
                                                  const float* __restrict__ gnp3,
                                                  float* __restrict__ out) {
    size_t i = (size_t)blockIdx.x * 256 + threadIdx.x;
    size_t bc = i / NNP;
    float sc = gnp3[bc * 2], sh = gnp3[bc * 2 + 1];
    out[i] = lrelu(fmaf(zbuf[i], sc, sh));
}

// ---------------------------------------------------------------------------
// Workspace layout
// ---------------------------------------------------------------------------
constexpr size_t SZ_IDX = (size_t)BB * NNP * KKN * sizeof(int);
constexpr size_t SZ_BNC = (size_t)BB * NNP * CC * sizeof(float);
constexpr size_t SZ_GNP = (size_t)BB * CC * 2 * sizeof(float);
constexpr size_t SZ_P1  = (size_t)BB * 64 * GG * 2 * sizeof(float);
constexpr size_t SZ_P23 = (size_t)BB * (NNP / 4) * GG * 2 * sizeof(float);
constexpr size_t OFF_IDX = 0;
constexpr size_t OFF_NF  = OFF_IDX + SZ_IDX;
constexpr size_t OFF_Q   = OFF_NF + SZ_BNC;
constexpr size_t OFF_K   = OFF_Q + SZ_BNC;
constexpr size_t OFF_V   = OFF_K + SZ_BNC;
constexpr size_t OFF_Z   = OFF_V + SZ_BNC;
constexpr size_t OFF_G1  = OFF_Z + SZ_BNC;
constexpr size_t OFF_G2  = OFF_G1 + SZ_GNP;
constexpr size_t OFF_G3  = OFF_G2 + SZ_GNP;
constexpr size_t OFF_P1  = OFF_G3 + SZ_GNP;
constexpr size_t OFF_P2  = OFF_P1 + SZ_P1;
constexpr size_t OFF_P3  = OFF_P2 + SZ_P23;
constexpr size_t OFF_T   = (OFF_P3 + SZ_P23 + 255) & ~(size_t)255;

extern "C" void kernel_launch(void* const* d_in, const int* in_sizes, int n_in,
                              void* d_out, int out_size, void* d_ws, size_t ws_size,
                              hipStream_t stream) {
    const float* xyz     = (const float*)d_in[0];
    const float* feat    = (const float*)d_in[1];
    const float* pre_w   = (const float*)d_in[2];
    const float* pre_b   = (const float*)d_in[3];
    const float* wq_w    = (const float*)d_in[4];
    const float* wq_b    = (const float*)d_in[5];
    const float* wk_w    = (const float*)d_in[6];
    const float* wk_b    = (const float*)d_in[7];
    const float* wv_w    = (const float*)d_in[8];
    const float* wv_b    = (const float*)d_in[9];
    const float* pos1_w  = (const float*)d_in[10];
    const float* pos1_b  = (const float*)d_in[11];
    const float* pos1_g  = (const float*)d_in[12];
    const float* pos1_be = (const float*)d_in[13];
    const float* pos2_w  = (const float*)d_in[14];
    const float* pos2_b  = (const float*)d_in[15];
    const float* att1_w  = (const float*)d_in[16];
    const float* att1_b  = (const float*)d_in[17];
    const float* att1_g  = (const float*)d_in[18];
    const float* att1_be = (const float*)d_in[19];
    const float* att2_w  = (const float*)d_in[20];
    const float* att2_b  = (const float*)d_in[21];
    const float* post_w  = (const float*)d_in[22];
    const float* post_b  = (const float*)d_in[23];
    const float* post_g  = (const float*)d_in[24];
    const float* post_be = (const float*)d_in[25];

    char* ws = (char*)d_ws;
    int*   idx   = (int*)(ws + OFF_IDX);
    float* nf    = (float*)(ws + OFF_NF);
    float* qb    = (float*)(ws + OFF_Q);
    float* kb    = (float*)(ws + OFF_K);
    float* vb    = (float*)(ws + OFF_V);
    float* zb    = (float*)(ws + OFF_Z);
    float* gnp1  = (float*)(ws + OFF_G1);
    float* gnp2  = (float*)(ws + OFF_G2);
    float* gnp3  = (float*)(ws + OFF_G3);
    float* part1 = (float*)(ws + OFF_P1);
    float* part2 = (float*)(ws + OFF_P2);
    float* part3 = (float*)(ws + OFF_P3);
    float* tbuf  = (float*)(ws + OFF_T);

    k_knn<<<BB * (NNP / 128), 128, 0, stream>>>(xyz, idx);
    k_pre_qkv<<<BB * (NNP / 64), 256, 0, stream>>>(feat, pre_w, pre_b, wq_w, wq_b,
                                                   wk_w, wk_b, wv_w, wv_b,
                                                   nf, qb, kb, vb);
    k_pos1_stats<<<BB * (NNP / 128), 256, 0, stream>>>(xyz, idx, pos1_w, pos1_b, part1);
    k_gn_reduce<<<BB * GG, 256, 0, stream>>>(part1, NNP / 128,
                                             (float)((size_t)CG * NNP * KKN),
                                             pos1_g, pos1_be, gnp1);
    k_mid<<<BB * (NNP / 4), 256, 0, stream>>>(xyz, idx, pos1_w, pos1_b, gnp1,
                                              pos2_w, pos2_b, att1_w, att1_b,
                                              qb, kb, tbuf, part2);
    k_gn_reduce<<<BB * GG, 256, 0, stream>>>(part2, NNP / 4,
                                             (float)((size_t)CG * NNP * KKN),
                                             att1_g, att1_be, gnp2);
    k_final<<<BB * (NNP / 4), 256, 0, stream>>>(idx, tbuf, gnp2, att2_w, att2_b,
                                                vb, nf, post_w, post_b, zb, part3);
    k_gn_reduce<<<BB * GG, 256, 0, stream>>>(part3, NNP / 4,
                                             (float)((size_t)CG * NNP),
                                             post_g, post_be, gnp3);
    k_finalize<<<(BB * CC * NNP) / 256, 256, 0, stream>>>(zb, gnp3, (float*)d_out);
}
